// HyperLSTM_12171937317003
// MI455X (gfx1250) — matrix operand order repaired and checked by reading, not compiled, not run on hardware
//
#include <hip/hip_runtime.h>
#include <hip/hip_bf16.h>
#include <math.h>

#define S_LEN 256
#define B_SZ  32
#define I_SZ  256
#define H_SZ  1024
#define HH_SZ 256
#define E_SZ  32
#define FB_C  1.0f

typedef float        v8f   __attribute__((ext_vector_type(8)));
typedef __bf16       v16bf __attribute__((ext_vector_type(16)));
typedef unsigned int u32x4 __attribute__((ext_vector_type(4)));

union Frag16 { u32x4 u[2]; v16bf h; };

// ---------------------------------------------------------------------------
// Fragment loaders (CDNA5 wave32 WMMA layouts, 16-bit operands)
//
// A (16x32, row-major, row contiguous along K):
//   lane L: m = L&15, half = L>>4
//   VGPR 0..3  hold K = half*8 + {0..7}      (contiguous 16B)
//   VGPR 4..7  hold K = 16 + half*8 + {0..7} (contiguous 16B)
// B (32x16 = W stored [N][K] row-major; column n of B = row n of W):
//   lane L: n = L&15, half = L>>4
//   VGPR 0..3 hold K = half*8 + {0..7}, VGPR 4..7 hold K = 16 + half*8 + {0..7}
// C/D (16x16 f32): lane L: n = L&15; VGPR v: m = v + 8*(L>>4)
// ---------------------------------------------------------------------------
__device__ __forceinline__ v16bf load_a_frag(const __bf16* base, int row0,
                                             int ldk, int k0, int lane) {
  int m = lane & 15, half = lane >> 4;
  const __bf16* r = base + (size_t)(row0 + m) * ldk + k0 + 8 * half;
  Frag16 f;
  f.u[0] = *(const u32x4*)(r);
  f.u[1] = *(const u32x4*)(r + 16);
  return f.h;
}

__device__ __forceinline__ v16bf load_b_frag(const __bf16* base, int n0,
                                             int ldk, int k0, int lane) {
  int n = lane & 15, half = lane >> 4;
  const __bf16* r = base + (size_t)(n0 + n) * ldk + k0 + 8 * half;
  Frag16 f;
  f.u[0] = *(const u32x4*)(r);
  f.u[1] = *(const u32x4*)(r + 16);
  return f.h;
}

#define WMMA_BF16(a, b, c) \
  __builtin_amdgcn_wmma_f32_16x16x32_bf16(false, (a), false, (b), (short)0, (c), false, false)

__device__ __forceinline__ float sigf(float x) { return 1.0f / (1.0f + __expf(-x)); }

// ---------------------------------------------------------------------------
// fp32 -> bf16 weight staging
// ---------------------------------------------------------------------------
__global__ void f32_to_bf16_kernel(const float* __restrict__ src,
                                   __bf16* __restrict__ dst, int n) {
  int i = blockIdx.x * blockDim.x + threadIdx.x;
  if (i < n) dst[i] = (__bf16)src[i];
}

// ---------------------------------------------------------------------------
// Per-step kernel 1: hyper-LSTM + z projections. Single block, 16 waves.
//   hg = xt@hwx^T + hh@hwh^T + hwb      (32 x 1024)    -> register fragments
//   hyper cell update (register-local)  -> hh (32 x 256), hc
//   z_q = hh@w_hz_q^T (+bias), q=x,h,b  (32 x 128 each) -> z_ws
// Wave w owns column group j = [w*16, w*16+16) of each of the 4 hyper gates,
// so i/f/j/o for one element sit in the same lane across the 4 accumulators.
// ---------------------------------------------------------------------------
__global__ __launch_bounds__(512) void hyper_step_kernel(
    const float* __restrict__ xt,
    const __bf16* __restrict__ hwx_bf, const __bf16* __restrict__ hwh_bf,
    const float* __restrict__ hwb,
    const __bf16* __restrict__ whz_bf,
    const float* __restrict__ b_hz_x, const float* __restrict__ b_hz_h,
    __bf16* __restrict__ x_bf, __bf16* __restrict__ hh_bf,
    float* __restrict__ hc_ws, float* __restrict__ z_ws) {
  __shared__ __align__(16) __bf16 x_lds[B_SZ * I_SZ];
  __shared__ __align__(16) __bf16 hh_lds[B_SZ * HH_SZ];

  const int tid  = threadIdx.x;
  const int lane = tid & 31;
  const int w    = tid >> 5;  // 0..15

  // Stage activations: xt -> bf16 (LDS + global for the main kernel), hh -> LDS
  for (int i = tid; i < B_SZ * I_SZ; i += 512) {
    __bf16 v = (__bf16)xt[i];
    x_lds[i] = v;
    x_bf[i]  = v;
  }
  for (int i = tid; i < B_SZ * HH_SZ; i += 512) hh_lds[i] = hh_bf[i];
  __syncthreads();

  // hg GEMM: 2 m-tiles x 4 gates per wave, K = 256 (x) + 256 (hh)
  v8f acc[2][4];
#pragma unroll
  for (int mt = 0; mt < 2; ++mt)
#pragma unroll
    for (int g = 0; g < 4; ++g) acc[mt][g] = (v8f){0, 0, 0, 0, 0, 0, 0, 0};

  const int jn0 = w * 16;
  for (int k0 = 0; k0 < I_SZ; k0 += 32) {
#pragma unroll
    for (int mt = 0; mt < 2; ++mt) {
      v16bf ax = load_a_frag(x_lds, mt * 16, I_SZ, k0, lane);
      v16bf ah = load_a_frag(hh_lds, mt * 16, HH_SZ, k0, lane);
#pragma unroll
      for (int g = 0; g < 4; ++g) {
        v16bf bx = load_b_frag(hwx_bf, g * HH_SZ + jn0, I_SZ, k0, lane);
        v16bf bh = load_b_frag(hwh_bf, g * HH_SZ + jn0, HH_SZ, k0, lane);
        acc[mt][g] = WMMA_BF16(ax, bx, acc[mt][g]);
        acc[mt][g] = WMMA_BF16(ah, bh, acc[mt][g]);
      }
    }
  }
  __syncthreads();  // all waves finished reading hh_lds

  // Hyper LSTM cell update, fragment-resident
  {
    const int n = lane & 15, half = lane >> 4;
    const int j = jn0 + n;
#pragma unroll
    for (int mt = 0; mt < 2; ++mt) {
#pragma unroll
      for (int v = 0; v < 8; ++v) {
        int   b   = mt * 16 + v + 8 * half;
        float hi  = acc[mt][0][v] + hwb[0 * HH_SZ + j];
        float hf  = acc[mt][1][v] + hwb[1 * HH_SZ + j];
        float hj  = acc[mt][2][v] + hwb[2 * HH_SZ + j];
        float ho  = acc[mt][3][v] + hwb[3 * HH_SZ + j];
        float hco = hc_ws[b * HH_SZ + j];
        float hcn = hco * sigf(hf + FB_C) + sigf(hi) * tanhf(hj);
        float hhn = tanhf(hcn) * sigf(ho);
        hc_ws[b * HH_SZ + j] = hcn;
        __bf16 hb            = (__bf16)hhn;
        hh_lds[b * HH_SZ + j] = hb;
        hh_bf[b * HH_SZ + j]  = hb;
      }
    }
  }
  __syncthreads();

  // z GEMMs: 3 projections x (2 m-tiles x 8 n-tiles) = 48 tile jobs, 3 per wave
  for (int jb = 0; jb < 3; ++jb) {
    int q  = (w * 3 + jb) >> 4;
    int r  = (w * 3 + jb) & 15;
    int mt = r >> 3;
    int n0 = (r & 7) * 16;
    v8f zacc = (v8f){0, 0, 0, 0, 0, 0, 0, 0};
    for (int k0 = 0; k0 < HH_SZ; k0 += 32) {
      v16bf a = load_a_frag(hh_lds, mt * 16, HH_SZ, k0, lane);
      v16bf b = load_b_frag(whz_bf + (size_t)q * (4 * E_SZ * HH_SZ), n0, HH_SZ, k0, lane);
      zacc = WMMA_BF16(a, b, zacc);
    }
    const int n = lane & 15, half = lane >> 4;
    int   col  = n0 + n;
    float bias = (q == 0) ? b_hz_x[col] : (q == 1) ? b_hz_h[col] : 0.0f;
#pragma unroll
    for (int v = 0; v < 8; ++v) {
      int m = mt * 16 + v + 8 * half;
      z_ws[(size_t)q * (B_SZ * 4 * E_SZ) + m * (4 * E_SZ) + col] = zacc[v] + bias;
    }
  }
}

// ---------------------------------------------------------------------------
// Per-step kernel 2: main gates + LSTM cell. 16 blocks x 8 waves.
// Block owns j in [bid*64, bid*64+64). Wave (mt, jt) keeps separate
// accumulators for gx and gh (they get different d_* scales) across all 4
// gates, so the full d-scaling + cell update is register-local per element.
// ---------------------------------------------------------------------------
__global__ __launch_bounds__(256) void main_step_kernel(
    const __bf16* __restrict__ x_bf, const __bf16* __restrict__ wx_bf,
    const __bf16* __restrict__ wh_bf, const __bf16* __restrict__ h_bf_in,
    const float* __restrict__ z_ws,
    const float* __restrict__ w_zd_x, const float* __restrict__ w_zd_h,
    const float* __restrict__ w_zd_b, const float* __restrict__ b_zd_b,
    float* __restrict__ c_ws, float* __restrict__ h_out,
    __bf16* __restrict__ h_bf_out) {
  const int lane  = threadIdx.x & 31;
  const int wv    = threadIdx.x >> 5;  // 0..7
  const int mt    = wv >> 2;           // batch tile
  const int jt    = wv & 3;            // 16-col group inside block
  const int jbase = blockIdx.x * 64 + jt * 16;

  v8f accX[4], accH[4];
#pragma unroll
  for (int g = 0; g < 4; ++g) {
    accX[g] = (v8f){0, 0, 0, 0, 0, 0, 0, 0};
    accH[g] = (v8f){0, 0, 0, 0, 0, 0, 0, 0};
  }

  for (int k0 = 0; k0 < I_SZ; k0 += 32) {
    v16bf a = load_a_frag(x_bf, mt * 16, I_SZ, k0, lane);
#pragma unroll
    for (int g = 0; g < 4; ++g) {
      v16bf b = load_b_frag(wx_bf, g * H_SZ + jbase, I_SZ, k0, lane);
      accX[g] = WMMA_BF16(a, b, accX[g]);
    }
  }
  for (int k0 = 0; k0 < H_SZ; k0 += 32) {
    v16bf a = load_a_frag(h_bf_in, mt * 16, H_SZ, k0, lane);
#pragma unroll
    for (int g = 0; g < 4; ++g) {
      v16bf b = load_b_frag(wh_bf, g * H_SZ + jbase, H_SZ, k0, lane);
      accH[g] = WMMA_BF16(a, b, accH[g]);
    }
  }

  // d_* scaling + LSTM cell update
  const int n = lane & 15, half = lane >> 4;
  const int j = jbase + n;
#pragma unroll
  for (int v = 0; v < 8; ++v) {
    int   b = mt * 16 + v + 8 * half;
    float gate[4];
#pragma unroll
    for (int g = 0; g < 4; ++g) {
      const float* zx = z_ws + 0 * (B_SZ * 4 * E_SZ) + b * (4 * E_SZ) + g * E_SZ;
      const float* zh = z_ws + 1 * (B_SZ * 4 * E_SZ) + b * (4 * E_SZ) + g * E_SZ;
      const float* zb = z_ws + 2 * (B_SZ * 4 * E_SZ) + b * (4 * E_SZ) + g * E_SZ;
      size_t woff = ((size_t)g * H_SZ + j) * E_SZ;
      float  dx = 0.f, dh = 0.f, db = 0.f;
#pragma unroll
      for (int e = 0; e < E_SZ; ++e) {
        dx += zx[e] * w_zd_x[woff + e];
        dh += zh[e] * w_zd_h[woff + e];
        db += zb[e] * w_zd_b[woff + e];
      }
      db += b_zd_b[g * H_SZ + j];
      gate[g] = dx * accX[g][v] + dh * accH[g][v] + db;
    }
    float co = c_ws[b * H_SZ + j];
    float cn = co * sigf(gate[1] + FB_C) + sigf(gate[0]) * tanhf(gate[2]);
    float hn = tanhf(cn) * sigf(gate[3]);
    c_ws[b * H_SZ + j]     = cn;
    h_out[b * H_SZ + j]    = hn;
    h_bf_out[b * H_SZ + j] = (__bf16)hn;
  }
}

// ---------------------------------------------------------------------------
extern "C" void kernel_launch(void* const* d_in, const int* in_sizes, int n_in,
                              void* d_out, int out_size, void* d_ws, size_t ws_size,
                              hipStream_t stream) {
  (void)in_sizes; (void)n_in; (void)out_size; (void)ws_size;
  const float* x      = (const float*)d_in[0];
  const float* wx     = (const float*)d_in[1];
  const float* wh     = (const float*)d_in[2];
  const float* hwx    = (const float*)d_in[3];
  const float* hwh    = (const float*)d_in[4];
  const float* hwb    = (const float*)d_in[5];
  const float* w_hz_x = (const float*)d_in[6];
  const float* b_hz_x = (const float*)d_in[7];
  const float* w_hz_h = (const float*)d_in[8];
  const float* b_hz_h = (const float*)d_in[9];
  const float* w_hz_b = (const float*)d_in[10];
  const float* w_zd_x = (const float*)d_in[11];
  const float* w_zd_h = (const float*)d_in[12];
  const float* w_zd_b = (const float*)d_in[13];
  const float* b_zd_b = (const float*)d_in[14];
  float*       h_out  = (float*)d_out;

  char*  ws  = (char*)d_ws;
  size_t off = 0;
  auto alloc = [&](size_t bytes) -> void* {
    void* p = ws + off;
    off += (bytes + 255) & ~(size_t)255;
    return p;
  };
  // bf16 weight staging (L2-resident working set, ~11 MB)
  __bf16* wx_bf  = (__bf16*)alloc((size_t)4 * H_SZ * I_SZ * 2);
  __bf16* wh_bf  = (__bf16*)alloc((size_t)4 * H_SZ * H_SZ * 2);
  __bf16* hwx_bf = (__bf16*)alloc((size_t)4 * HH_SZ * I_SZ * 2);
  __bf16* hwh_bf = (__bf16*)alloc((size_t)4 * HH_SZ * HH_SZ * 2);
  __bf16* whz_bf = (__bf16*)alloc((size_t)3 * 4 * E_SZ * HH_SZ * 2);
  __bf16* x_bf   = (__bf16*)alloc((size_t)B_SZ * I_SZ * 2);
  float*  z_ws   = (float*)alloc((size_t)3 * B_SZ * 4 * E_SZ * 4);
  // state (zero-initialized as one contiguous memset region)
  size_t  state_begin = off;
  float*  c_ws  = (float*)alloc((size_t)B_SZ * H_SZ * 4);
  float*  hc_ws = (float*)alloc((size_t)B_SZ * HH_SZ * 4);
  __bf16* hbf0  = (__bf16*)alloc((size_t)B_SZ * H_SZ * 2);
  __bf16* hbf1  = (__bf16*)alloc((size_t)B_SZ * H_SZ * 2);
  __bf16* hh_bf = (__bf16*)alloc((size_t)B_SZ * HH_SZ * 2);
  size_t  state_bytes = off - state_begin;

  hipMemsetAsync(ws + state_begin, 0, state_bytes, stream);

  // Stage weights as bf16 (once per launch)
  auto cvt = [&](const float* s, __bf16* d, int n) {
    f32_to_bf16_kernel<<<(n + 255) / 256, 256, 0, stream>>>(s, d, n);
  };
  cvt(wx,     wx_bf,  4 * H_SZ * I_SZ);
  cvt(wh,     wh_bf,  4 * H_SZ * H_SZ);
  cvt(hwx,    hwx_bf, 4 * HH_SZ * I_SZ);
  cvt(hwh,    hwh_bf, 4 * HH_SZ * HH_SZ);
  cvt(w_hz_x, whz_bf + 0 * (4 * E_SZ * HH_SZ), 4 * E_SZ * HH_SZ);
  cvt(w_hz_h, whz_bf + 1 * (4 * E_SZ * HH_SZ), 4 * E_SZ * HH_SZ);
  cvt(w_hz_b, whz_bf + 2 * (4 * E_SZ * HH_SZ), 4 * E_SZ * HH_SZ);

  // Sequential scan: 2 kernels per step, double-buffered h_bf
  for (int t = 0; t < S_LEN; ++t) {
    const __bf16* hin  = (t & 1) ? hbf1 : hbf0;
    __bf16*       hout = (t & 1) ? hbf0 : hbf1;
    hyper_step_kernel<<<dim3(1), dim3(512), 0, stream>>>(
        x + (size_t)t * B_SZ * I_SZ, hwx_bf, hwh_bf, hwb, whz_bf,
        b_hz_x, b_hz_h, x_bf, hh_bf, hc_ws, z_ws);
    main_step_kernel<<<dim3(16), dim3(256), 0, stream>>>(
        x_bf, wx_bf, wh_bf, hin, z_ws, w_zd_x, w_zd_h, w_zd_b, b_zd_b,
        c_ws, h_out, hout);
  }
}
